// XFiAggregator_48868137894521
// MI455X (gfx1250) — matrix operand order repaired and checked by reading, not compiled, not run on hardware
//
#include <hip/hip_runtime.h>
#include <hip/hip_bf16.h>
#include <math.h>

// ---------------------------------------------------------------------------
// Model constants
// ---------------------------------------------------------------------------
#define BATCH   64
#define DIM     512
#define NHEAD   8
#define DHEAD   64
#define NMOD    4
#define NPOINT  512
#define NPTS    65536
#define NTOK    128          // NMOD * 32
#define HID     1024

typedef __attribute__((ext_vector_type(16))) __bf16        v16bf;
typedef __attribute__((ext_vector_type(8)))  float         v8f;
typedef __attribute__((ext_vector_type(8)))  unsigned int  v8u;

// ---------------------------------------------------------------------------
// Wave32 / block reduction helpers
// ---------------------------------------------------------------------------
__device__ __forceinline__ float waveReduceSum(float v) {
    #pragma unroll
    for (int off = 16; off; off >>= 1) v += __shfl_xor(v, off, 32);
    return v;
}

__device__ __forceinline__ float blockReduceSum(float v, float* red) {
    v = waveReduceSum(v);
    const int w = threadIdx.x >> 5, lane = threadIdx.x & 31;
    if (lane == 0) red[w] = v;
    __syncthreads();
    const int nw = blockDim.x >> 5;
    float t = (threadIdx.x < (unsigned)nw) ? red[threadIdx.x] : 0.f;
    if (threadIdx.x < 32) t = waveReduceSum(t);
    if (threadIdx.x == 0) red[0] = t;
    __syncthreads();
    float r = red[0];
    __syncthreads();
    return r;
}

__device__ __forceinline__ unsigned pack_bf16x2(float a, float b) {
    union { __bf16 h[2]; unsigned u; } p;
    p.h[0] = (__bf16)a;      // compiler fuses pair into v_cvt_pk_bf16_f32
    p.h[1] = (__bf16)b;
    return p.u;
}

// ---------------------------------------------------------------------------
// 1) Farthest point sampling + gather.
//    One workgroup per batch.  dist[65536] (256 KB) lives in dynamic LDS for
//    all 512 rounds (WGP has 320 KB); xyz (48 MB) stays L2-resident.
// ---------------------------------------------------------------------------
__global__ __launch_bounds__(1024)
void fps_gather_kernel(const float* __restrict__ xyz, float* __restrict__ out_xyz)
{
    extern __shared__ float smem[];
    float* dist = smem;                 // NPTS floats
    float* rv   = smem + NPTS;          // 32 per-wave best values
    int*   ri   = (int*)(rv + 32);      // 32 per-wave best indices
    int*   ctrl = ri + 32;              // ctrl[0] = current farthest index

    const int b = blockIdx.x;
    const float* xb = xyz + (size_t)b * NPTS * 3;

    for (int p = threadIdx.x; p < NPTS; p += blockDim.x) dist[p] = 1e10f;
    if (threadIdx.x == 0) ctrl[0] = 0;
    __syncthreads();

    for (int j = 0; j < NPOINT; ++j) {
        const int far = ctrl[0];
        const float fx = xb[far * 3 + 0];
        const float fy = xb[far * 3 + 1];
        const float fz = xb[far * 3 + 2];
        if (threadIdx.x == 0) {
            float* o = out_xyz + ((size_t)b * NPOINT + j) * 3;
            o[0] = fx; o[1] = fy; o[2] = fz;        // scan emits carry-before-update
        }
        float bestv = -1.f; int besti = 0;
        for (int p = threadIdx.x; p < NPTS; p += blockDim.x) {
            __builtin_prefetch(&xb[(size_t)(p + blockDim.x) * 3], 0, 0); // global_prefetch_b8
            const float3 pt = *(const float3*)(xb + (size_t)p * 3);      // global_load_b96
            const float dx = pt.x - fx, dy = pt.y - fy, dz = pt.z - fz;
            float dm = fminf(dist[p], dx * dx + dy * dy + dz * dz);
            dist[p] = dm;
            if (dm > bestv) { bestv = dm; besti = p; }
        }
        // wave32 argmax (ties -> lowest index, matching jnp.argmax)
        #pragma unroll
        for (int off = 16; off; off >>= 1) {
            const float ov = __shfl_xor(bestv, off, 32);
            const int   oi = __shfl_xor(besti, off, 32);
            if (ov > bestv || (ov == bestv && oi < besti)) { bestv = ov; besti = oi; }
        }
        const int w = threadIdx.x >> 5;
        if ((threadIdx.x & 31) == 0) { rv[w] = bestv; ri[w] = besti; }
        __syncthreads();
        if (threadIdx.x == 0) {
            float bv = rv[0]; int bi = ri[0];
            const int nw = blockDim.x >> 5;
            for (int t = 1; t < nw; ++t)
                if (rv[t] > bv || (rv[t] == bv && ri[t] < bi)) { bv = rv[t]; bi = ri[t]; }
            ctrl[0] = bi;
        }
        __syncthreads();
    }
}

// ---------------------------------------------------------------------------
// 2) Generic batched GEMM:  C[b] = act(A[b] @ W + bias)
//    A: [batch(strideA), M, K] f32, W: [K, N] f32 shared, C: [batch, M, N].
//    bf16 WMMA: 32x64 block tile, 8 waves -> 2x4 grid of 16x16 tiles, K-step
//    32.  LDS holds bf16 pairs in WMMA-fragment order so each fragment is two
//    ds_load_b128; global staging is b128/b64 and software-pipelined past the
//    WMMA.  ACT: 0 = none, 1 = exact GELU, 2 = ReLU.
// ---------------------------------------------------------------------------
template<int ACT>
__global__ __launch_bounds__(256)
void gemm_bias_act_kernel(const float* __restrict__ A, const float* __restrict__ W,
                          const float* __restrict__ bias, float* __restrict__ C,
                          int M, int N, int K, long strideA, long strideC)
{
    // [row][kpair]: dword packs bf16 (k=2p, 2p+1).  Row stride 20 dwords:
    // 16B-aligned rows, gcd(20,64)=4 but row*20 mod 64 spans 16 distinct
    // values over 16 rows -> conflict-free b128 fragment loads.
    __shared__ __align__(16) unsigned As32[32][20];
    __shared__ __align__(16) unsigned Wt[64][20];     // transposed: [col][kpair]

    const int b  = blockIdx.z;
    const int m0 = blockIdx.y * 32;
    const int n0 = blockIdx.x * 64;
    const float* Ab = A + (size_t)b * strideA;
    float*       Cb = C + (size_t)b * strideC;

    const int tid  = threadIdx.x;
    const int lane = tid & 31;
    const int wv   = tid >> 5;
    const int wm   = wv >> 2;       // 0..1
    const int wn   = wv & 3;        // 0..3

    // A staging: thread -> (row = tid/8, 4-float chunk = tid%8)
    const int am = tid >> 3, ac = tid & 7;
    // W staging: thread handles two 2x2 blocks; bid -> (kpair = bid%16, npair = bid/16)
    float4 aReg;
    float2 w0[2], w1[2];

    auto fetch = [&](int k0) {
        aReg = *(const float4*)&Ab[(size_t)(m0 + am) * K + k0 + ac * 4];   // b128
        #pragma unroll
        for (int j = 0; j < 2; ++j) {
            const int bid = tid + j * 256;
            const int kp = bid & 15, np = bid >> 4;
            const float* wp = &W[(size_t)(k0 + 2 * kp) * N + n0 + 2 * np];
            w0[j] = *(const float2*)wp;          // b64, row k=2kp
            w1[j] = *(const float2*)(wp + N);    // b64, row k=2kp+1
        }
    };
    auto stage = [&]() {
        As32[am][ac * 2 + 0] = pack_bf16x2(aReg.x, aReg.y);
        As32[am][ac * 2 + 1] = pack_bf16x2(aReg.z, aReg.w);  // ds_store_b64
        #pragma unroll
        for (int j = 0; j < 2; ++j) {
            const int bid = tid + j * 256;
            const int kp = bid & 15, np = bid >> 4;
            Wt[2 * np + 0][kp] = pack_bf16x2(w0[j].x, w1[j].x);
            Wt[2 * np + 1][kp] = pack_bf16x2(w0[j].y, w1[j].y);
        }
    };

    v8f acc = {};
    const int mr = (lane & 15) + wm * 16;
    const int nc = (lane & 15) + wn * 16;
    const int pA = (lane < 16) ? 0 : 4;   // A pair base (ISA 16-bit 16x32 layout)
    const int pB = (lane < 16) ? 0 : 4;   // B pair base (ISA 16-bit 16x32 layout)

    fetch(0);
    for (int k0 = 0; k0 < K; k0 += 32) {
        stage();
        __syncthreads();
        if (k0 + 32 < K) fetch(k0 + 32);   // overlap next tile with WMMA

        union { v8u u; v16bf h; } af, bf;
        const uint4 alo = *(const uint4*)&As32[mr][pA];        // pairs pA..pA+3
        const uint4 ahi = *(const uint4*)&As32[mr][pA + 8];    // pairs pA+8..pA+11
        af.u[0] = alo.x; af.u[1] = alo.y; af.u[2] = alo.z; af.u[3] = alo.w;
        af.u[4] = ahi.x; af.u[5] = ahi.y; af.u[6] = ahi.z; af.u[7] = ahi.w;
        const uint4 blo = *(const uint4*)&Wt[nc][pB];          // pairs pB..pB+3
        const uint4 bhi = *(const uint4*)&Wt[nc][pB + 8];      // pairs pB+8..pB+11
        bf.u[0] = blo.x; bf.u[1] = blo.y; bf.u[2] = blo.z; bf.u[3] = blo.w;
        bf.u[4] = bhi.x; bf.u[5] = bhi.y; bf.u[6] = bhi.z; bf.u[7] = bhi.w;

        acc = __builtin_amdgcn_wmma_f32_16x16x32_bf16(
                  false, af.h, false, bf.h, (short)0, acc, false, false);
        __syncthreads();
    }

    #pragma unroll
    for (int r = 0; r < 8; ++r) {
        const int m = m0 + wm * 16 + r + ((lane < 16) ? 0 : 8);
        const int n = n0 + wn * 16 + (lane & 15);
        float v = acc[r];
        if (bias) v += bias[n];
        if (ACT == 1) v = 0.5f * v * (1.0f + erff(v * 0.70710678118654752f));
        if (ACT == 2) v = fmaxf(v, 0.f);
        Cb[(size_t)m * N + n] = v;
    }
}

// ---------------------------------------------------------------------------
// 3) Attention: one wave32 per query row, lanes split the 64-dim head.
//    prescale applied to logits before softmax; postscale after normalize.
// ---------------------------------------------------------------------------
__global__ __launch_bounds__(256)
void attention_kernel(const float* __restrict__ Qb, const float* __restrict__ Kb,
                      const float* __restrict__ Vb, float* __restrict__ O,
                      int n, long q_sb, long q_sh, long q_sn, int q_sd,
                      long kv_sb, long kv_sh, long kv_sn, int kv_sd,
                      long o_sb, float prescale, float postscale)
{
    __shared__ float sc[8][132];
    const int b = blockIdx.z, h = blockIdx.y;
    const int w = threadIdx.x >> 5, lane = threadIdx.x & 31;
    const int row = blockIdx.x * 8 + w;

    const float* q  = Qb + (long)b * q_sb + (long)h * q_sh + (long)row * q_sn;
    const float* Kp = Kb + (long)b * kv_sb + (long)h * kv_sh;
    const float* Vp = Vb + (long)b * kv_sb + (long)h * kv_sh;

    const int d0 = lane * 2;
    const float q0 = q[(long)d0 * q_sd];
    const float q1 = q[(long)(d0 + 1) * q_sd];

    float mx = -1e30f;
    for (int k = 0; k < n; ++k) {
        const float* kr = Kp + (long)k * kv_sn;
        float p = q0 * kr[(long)d0 * kv_sd] + q1 * kr[(long)(d0 + 1) * kv_sd];
        p = waveReduceSum(p) * prescale;
        if (lane == 0) sc[w][k] = p;
        mx = fmaxf(mx, p);
    }
    __syncthreads();

    float sum = 0.f;
    for (int k = lane; k < n; k += 32) {
        const float e = expf(sc[w][k] - mx);
        sc[w][k] = e;
        sum += e;
    }
    sum = waveReduceSum(sum);
    const float inv = postscale / sum;
    __syncthreads();

    float o0 = 0.f, o1 = 0.f;
    for (int k = 0; k < n; ++k) {
        const float s = sc[w][k] * inv;
        const float* vr = Vp + (long)k * kv_sn;
        o0 += s * vr[(long)d0 * kv_sd];
        o1 += s * vr[(long)(d0 + 1) * kv_sd];
    }
    float* op = O + (long)b * o_sb + (long)row * DIM + h * DHEAD;
    op[d0] = o0; op[d0 + 1] = o1;
}

// ---------------------------------------------------------------------------
// 4) LayerNorm over last dim D (one block per row); optional fused ReLU.
// ---------------------------------------------------------------------------
__global__ __launch_bounds__(256)
void ln_kernel(const float* __restrict__ x, const float* __restrict__ g,
               const float* __restrict__ bb, float* __restrict__ y,
               int D, int relu_flag)
{
    __shared__ float red[32];
    __shared__ float s_mean, s_rstd;
    const long row = blockIdx.x;
    const float* xr = x + row * (long)D;

    float s = 0.f;
    for (int i = threadIdx.x; i < D; i += blockDim.x) s += xr[i];
    s = blockReduceSum(s, red);
    if (threadIdx.x == 0) s_mean = s / (float)D;
    __syncthreads();
    const float m = s_mean;

    float v = 0.f;
    for (int i = threadIdx.x; i < D; i += blockDim.x) {
        const float t = xr[i] - m; v += t * t;
    }
    v = blockReduceSum(v, red);
    if (threadIdx.x == 0) s_rstd = rsqrtf(v / (float)D + 1e-5f);
    __syncthreads();
    const float rs = s_rstd;

    for (int i = threadIdx.x; i < D; i += blockDim.x) {
        float o = (xr[i] - m) * rs * g[i] + bb[i];
        if (relu_flag) o = fmaxf(o, 0.f);
        y[row * (long)D + i] = o;
    }
}

// ---------------------------------------------------------------------------
// 5) pool32: [B, n, 512] -> [B, 32, 512] mean over groups of n/32
// ---------------------------------------------------------------------------
__global__ void pool32_kernel(const float* __restrict__ x, float* __restrict__ y,
                              int n, long total)
{
    const long idx = (long)blockIdx.x * blockDim.x + threadIdx.x;
    if (idx >= total) return;
    const int d = (int)(idx & 511);
    const int i = (int)((idx >> 9) & 31);
    const long b = idx >> 14;
    const int gsz = n / 32;
    const float* xr = x + ((long)b * n + (long)i * gsz) * DIM + d;
    float s = 0.f;
    for (int j = 0; j < gsz; ++j) s += xr[(long)j * DIM];
    y[idx] = s / (float)gsz;
}

// ---------------------------------------------------------------------------
// 6) elementwise add / scatter-add into concat slot (float4-wide)
// ---------------------------------------------------------------------------
__global__ void add_kernel(const float4* __restrict__ a, const float4* __restrict__ b,
                           float4* __restrict__ o, long n4)
{
    const long idx = (long)blockIdx.x * blockDim.x + threadIdx.x;
    if (idx < n4) {
        const float4 x = a[idx], y = b[idx];
        o[idx] = make_float4(x.x + y.x, x.y + y.y, x.z + y.z, x.w + y.w);
    }
}

__global__ void add_scatter_kernel(const float4* __restrict__ a, const float4* __restrict__ b,
                                   float4* __restrict__ o, int d4, int rows,
                                   long o_bstride4, long o_off4, long n4)
{
    const long idx = (long)blockIdx.x * blockDim.x + threadIdx.x;
    if (idx >= n4) return;
    const int col = (int)(idx % d4);
    const long t = idx / d4;
    const int row = (int)(t % rows);
    const long bb = t / rows;
    const float4 x = a[idx], y = b[idx];
    o[bb * o_bstride4 + o_off4 + (long)row * d4 + col] =
        make_float4(x.x + y.x, x.y + y.y, x.z + y.z, x.w + y.w);
}

// ---------------------------------------------------------------------------
// Host-side orchestration
// ---------------------------------------------------------------------------
static inline void launch_gemm(const float* A, const float* W, const float* bias,
                               float* C, int M, int N, int K, long strideA,
                               int act, hipStream_t stream)
{
    dim3 grid(N / 64, M / 32, BATCH);
    const long strideC = (long)M * N;
    if (act == 1)
        gemm_bias_act_kernel<1><<<grid, 256, 0, stream>>>(A, W, bias, C, M, N, K, strideA, strideC);
    else if (act == 2)
        gemm_bias_act_kernel<2><<<grid, 256, 0, stream>>>(A, W, bias, C, M, N, K, strideA, strideC);
    else
        gemm_bias_act_kernel<0><<<grid, 256, 0, stream>>>(A, W, bias, C, M, N, K, strideA, strideC);
}

static inline void launch_ln(const float* x, const float* g, const float* b,
                             float* y, long rows, int D, int relu, hipStream_t stream)
{
    ln_kernel<<<dim3((unsigned)rows), 256, 0, stream>>>(x, g, b, y, D, relu);
}

static inline void launch_add(const float* a, const float* b, float* o, long n,
                              hipStream_t stream)
{
    const long n4 = n / 4;
    add_kernel<<<dim3((unsigned)((n4 + 255) / 256)), 256, 0, stream>>>(
        (const float4*)a, (const float4*)b, (float4*)o, n4);
}

extern "C" void kernel_launch(void* const* d_in, const int* in_sizes, int n_in,
                              void* d_out, int out_size, void* d_ws, size_t ws_size,
                              hipStream_t stream)
{
    (void)in_sizes; (void)n_in; (void)out_size; (void)ws_size;

    // ---- inputs (JAX pytree flatten order: dict keys sorted) ----
    const float* feature = (const float*)d_in[0];   // [64,128,512]
    const float* xyz     = (const float*)d_in[1];   // [64,65536,3]
    // cmt: Wp, Wqkv, bp, bqkv, ffn{W1,W2,b,b1,b2,g}
    const float* cWp   = (const float*)d_in[2];
    const float* cWqkv = (const float*)d_in[3];
    const float* cbp   = (const float*)d_in[4];
    const float* cbqkv = (const float*)d_in[5];
    const float* cfW1  = (const float*)d_in[6];
    const float* cfW2  = (const float*)d_in[7];
    const float* cfb   = (const float*)d_in[8];
    const float* cfb1  = (const float*)d_in[9];
    const float* cfb2  = (const float*)d_in[10];
    const float* cfg   = (const float*)d_in[11];
    // fus[i]: Wo, beo, bo, ffn{W1,W2,b,b1,b2,g}, go   (base 12 + 10*i)
    // kv[i]:  W1, W2, Wk, Wv, b1, b2, be1, bn, g1, gn (base 52 + 10*i)

    float* out_fused = (float*)d_out;                          // [64,32,512]
    float* out_xyz   = (float*)d_out + (long)BATCH * 32 * DIM; // [64,512,3]

    // ---- workspace carve (floats) ----
    float* ws = (float*)d_ws;
    long off = 0;
    auto alloc = [&](long nelem) { float* p = ws + off; off += nelem; return p; };
    float* S_qkv    = alloc((long)BATCH * NTOK * 3 * DIM);   // 12.6M
    float* S_att    = alloc((long)BATCH * NTOK * DIM);       // 4.2M
    float* S_tmp    = alloc((long)BATCH * NTOK * DIM);       // 4.2M
    float* S_a      = alloc((long)BATCH * 32 * DIM);
    float* S_b      = alloc((long)BATCH * 32 * DIM);
    float* S_c      = alloc((long)BATCH * 32 * DIM);
    float* S_d      = alloc((long)BATCH * 32 * DIM);
    float* S_h      = alloc((long)BATCH * 32 * HID);         // 2.1M
    float* S_e      = alloc((long)BATCH * 32 * DIM);
    float* femb     = alloc((long)BATCH * 32 * DIM);
    float* femb_cat = alloc((long)BATCH * NTOK * DIM);       // 4.2M
    float* kbuf     = alloc((long)NMOD * BATCH * 32 * DIM);  // 4.2M
    float* vbuf     = alloc((long)NMOD * BATCH * 32 * DIM);  // 4.2M

    const long KVSTRIDE = (long)BATCH * 32 * DIM;
    const long NE32     = (long)BATCH * 32 * DIM;            // 1,048,576
    const float inv_sqrt_dim = 1.0f / sqrtf((float)DIM);

    // ---- 1) FPS + gather -> out_xyz ----
    {
        const size_t shmem = (size_t)NPTS * 4 + 32 * 4 + 32 * 4 + 16;
        fps_gather_kernel<<<BATCH, 1024, shmem, stream>>>(xyz, out_xyz);
    }

    // ---- 2) kvproj per modality ----
    for (int i = 0; i < NMOD; ++i) {
        const int base = 52 + 10 * i;
        const float* W1  = (const float*)d_in[base + 0];
        const float* W2  = (const float*)d_in[base + 1];
        const float* Wk  = (const float*)d_in[base + 2];
        const float* Wv  = (const float*)d_in[base + 3];
        const float* b1  = (const float*)d_in[base + 4];
        const float* b2  = (const float*)d_in[base + 5];
        const float* be1 = (const float*)d_in[base + 6];
        const float* bn  = (const float*)d_in[base + 7];
        const float* g1  = (const float*)d_in[base + 8];
        const float* gn  = (const float*)d_in[base + 9];

        const float* chunk = feature + (long)i * 32 * DIM;   // batch stride 128*512
        launch_gemm(chunk, W1, b1, S_h, 32, HID, DIM, (long)NTOK * DIM, 0, stream);
        launch_ln(S_h, g1, be1, S_qkv, (long)BATCH * 32, HID, /*relu*/1, stream);
        launch_gemm(S_qkv, W2, b2, S_att, 32, DIM, HID, (long)32 * HID, 0, stream);
        launch_ln(S_att, gn, bn, S_tmp, (long)BATCH * 32, DIM, 0, stream);
        launch_gemm(S_tmp, Wk, nullptr, kbuf + i * KVSTRIDE, 32, DIM, DIM, (long)32 * DIM, 0, stream);
        launch_gemm(S_tmp, Wv, nullptr, vbuf + i * KVSTRIDE, 32, DIM, DIM, (long)32 * DIM, 0, stream);
    }

    // ---- cmt helper ----
    auto run_cmt = [&](const float* x, float* out) {
        // qkv = x @ Wqkv + b   ([B,128,1536], packed (h d 3))
        launch_gemm(x, cWqkv, cbqkv, S_qkv, NTOK, 3 * DIM, DIM, (long)NTOK * DIM, 0, stream);
        // attention (post-softmax scale 1/sqrt(512))
        attention_kernel<<<dim3(NTOK / 8, NHEAD, BATCH), 256, 0, stream>>>(
            S_qkv + 0, S_qkv + 1, S_qkv + 2, S_att, NTOK,
            (long)NTOK * 3 * DIM, (long)3 * DHEAD, (long)3 * DIM, 3,
            (long)NTOK * 3 * DIM, (long)3 * DHEAD, (long)3 * DIM, 3,
            (long)NTOK * DIM, 1.0f, inv_sqrt_dim);
        // o @ Wp + bp
        launch_gemm(S_att, cWp, cbp, S_tmp, NTOK, DIM, DIM, (long)NTOK * DIM, 0, stream);
        // pool + residual pool
        pool32_kernel<<<dim3((unsigned)((NE32 + 255) / 256)), 256, 0, stream>>>(S_tmp, S_a, NTOK, NE32);
        pool32_kernel<<<dim3((unsigned)((NE32 + 255) / 256)), 256, 0, stream>>>(x, S_b, NTOK, NE32);
        launch_add(S_a, S_b, S_c, NE32, stream);             // y
        // ffw(y) + y
        launch_ln(S_c, cfg, cfb, S_d, (long)BATCH * 32, DIM, 0, stream);
        launch_gemm(S_d, cfW1, cfb1, S_h, 32, HID, DIM, (long)32 * DIM, /*gelu*/1, stream);
        launch_gemm(S_h, cfW2, cfb2, S_e, 32, DIM, HID, (long)32 * HID, 0, stream);
        launch_add(S_e, S_c, out, NE32, stream);
    };

    // ---- 3) femb = cmt(feature) ----
    run_cmt(feature, femb);

    // ---- 4) depth loop ----
    for (int depth = 0; depth < 2; ++depth) {
        for (int i = 0; i < NMOD; ++i) {
            const int base = 12 + 10 * i;
            const float* Wo  = (const float*)d_in[base + 0];
            const float* beo = (const float*)d_in[base + 1];
            const float* bo  = (const float*)d_in[base + 2];
            const float* fW1 = (const float*)d_in[base + 3];
            const float* fW2 = (const float*)d_in[base + 4];
            const float* fb  = (const float*)d_in[base + 5];
            const float* fb1 = (const float*)d_in[base + 6];
            const float* fb2 = (const float*)d_in[base + 7];
            const float* fg  = (const float*)d_in[base + 8];
            const float* go  = (const float*)d_in[base + 9];

            // cross attention (pre-softmax scale DH^-0.5)
            attention_kernel<<<dim3(32 / 8, NHEAD, BATCH), 256, 0, stream>>>(
                femb, kbuf + i * KVSTRIDE, vbuf + i * KVSTRIDE, S_att, 32,
                (long)32 * DIM, (long)DHEAD, (long)DIM, 1,
                (long)32 * DIM, (long)DHEAD, (long)DIM, 1,
                (long)32 * DIM, 0.125f, 1.0f);
            launch_gemm(S_att, Wo, bo, S_tmp, 32, DIM, DIM, (long)32 * DIM, 0, stream);
            launch_ln(S_tmp, go, beo, S_d, (long)BATCH * 32, DIM, 0, stream);
            launch_add(S_d, femb, S_c, NE32, stream);        // x = o + femb
            // ffw(x) + x -> concat slot i of femb_cat
            launch_ln(S_c, fg, fb, S_d, (long)BATCH * 32, DIM, 0, stream);
            launch_gemm(S_d, fW1, fb1, S_h, 32, HID, DIM, (long)32 * DIM, 1, stream);
            launch_gemm(S_h, fW2, fb2, S_e, 32, DIM, HID, (long)32 * HID, 0, stream);
            add_scatter_kernel<<<dim3((unsigned)((NE32 / 4 + 255) / 256)), 256, 0, stream>>>(
                (const float4*)S_e, (const float4*)S_c, (float4*)femb_cat,
                DIM / 4, 32, (long)NTOK * DIM / 4, (long)i * 32 * DIM / 4, NE32 / 4);
        }
        run_cmt(femb_cat, (depth == 1) ? out_fused : femb);
    }
}
